// MoeLayer_66769561584067
// MI455X (gfx1250) — matrix operand order repaired and checked by reading, not compiled, not run on hardware
//
#include <hip/hip_runtime.h>
#include <hip/hip_bf16.h>

// Problem sizes (match reference)
static constexpr int Tt = 2048;
static constexpr int Dd = 512;
static constexpr int Ff = 2048;
static constexpr int Ee = 8;

static constexpr int WP = 40;   // transposed weight tile row stride (ushorts): 32 k + 8 pad
                                // 80B rows -> 16B aligned, bank-spread (20 dwords)

typedef __attribute__((ext_vector_type(16))) __bf16 v16bf;
typedef __attribute__((ext_vector_type(8)))  float  v8f;

union FragB {
    v16bf v;
    uint4 q[2];
};

__device__ __forceinline__ unsigned short f2bf(float f) {
    unsigned u = __float_as_uint(f);
    unsigned r = u + 0x7FFFu + ((u >> 16) & 1u);   // RNE
    return (unsigned short)(r >> 16);
}

__device__ __forceinline__ unsigned pkbf(float a, float b) {
#if __has_builtin(__builtin_amdgcn_cvt_pk_bf16_f32)
    typedef __attribute__((ext_vector_type(2))) __bf16 v2bf;
    union { v2bf v; unsigned u; } p;
    p.v = __builtin_amdgcn_cvt_pk_bf16_f32(a, b);
    return p.u;
#else
    return (unsigned)f2bf(a) | ((unsigned)f2bf(b) << 16);
#endif
}

// A-fragment (16x32 bf16): per lane two 16B runs, 16 elements apart in K
__device__ __forceinline__ v16bf load_frag_a(const unsigned short* p) {
    FragB f;
    f.q[0] = *(const uint4*)(p);
    f.q[1] = *(const uint4*)(p + 16);
    return f.v;
}
// B-fragment (32x16 bf16) from a transposed [n][k] tile: per lane two 16B runs, 16 elements apart in K
__device__ __forceinline__ v16bf load_frag_b(const unsigned short* p) {
    FragB f;
    f.q[0] = *(const uint4*)(p);
    f.q[1] = *(const uint4*)(p + 16);
    return f.v;
}

__device__ __forceinline__ v8f vzero() {
    v8f z = {0.f, 0.f, 0.f, 0.f, 0.f, 0.f, 0.f, 0.f};
    return z;
}

// ---------------------------------------------------------------------------
__global__ void moe_zero(int* counts) {
    if (threadIdx.x < Ee) counts[threadIdx.x] = 0;
}

// ---------------------------------------------------------------------------
// Routing: one wave32 per token; keep the max-index expert of the top-2
// (reference's overwrite semantics) with its softmax weight.
// ---------------------------------------------------------------------------
__global__ __launch_bounds__(256) void moe_route(
    const float* __restrict__ x, const float* __restrict__ Wg,
    int* __restrict__ counts, int* __restrict__ list, float* __restrict__ wlist) {

    const int tokid = (blockIdx.x * blockDim.x + threadIdx.x) >> 5;
    const int lane  = threadIdx.x & 31;
    if (tokid >= Tt) return;

    const float* xr = x + (unsigned)tokid * (unsigned)Dd;
    float acc[Ee];
#pragma unroll
    for (int e = 0; e < Ee; ++e) acc[e] = 0.0f;

    for (int d = lane; d < Dd; d += 32) {
        float xv = xr[d];
        const float4* wr = (const float4*)(Wg + (unsigned)d * (unsigned)Ee);
        float4 w0 = wr[0];
        float4 w1 = wr[1];
        acc[0] += xv * w0.x; acc[1] += xv * w0.y;
        acc[2] += xv * w0.z; acc[3] += xv * w0.w;
        acc[4] += xv * w1.x; acc[5] += xv * w1.y;
        acc[6] += xv * w1.z; acc[7] += xv * w1.w;
    }
#pragma unroll
    for (int e = 0; e < Ee; ++e) {
#pragma unroll
        for (int off = 16; off > 0; off >>= 1)
            acc[e] += __shfl_xor(acc[e], off, 32);
    }

    if (lane == 0) {
        int i1 = 0; float v1 = acc[0];
#pragma unroll
        for (int e = 1; e < Ee; ++e) if (acc[e] > v1) { v1 = acc[e]; i1 = e; }
        int i2 = -1; float v2 = -3.4e38f;
#pragma unroll
        for (int e = 0; e < Ee; ++e) if (e != i1 && acc[e] > v2) { v2 = acc[e]; i2 = e; }
        float e2  = __expf(v2 - v1);
        float inv = 1.0f / (1.0f + e2);
        int ef; float wf;
        if (i1 > i2) { ef = i1; wf = inv; } else { ef = i2; wf = e2 * inv; }
        int slot = atomicAdd(&counts[ef], 1);
        list[ef * Tt + slot]  = tokid;
        wlist[ef * Tt + slot] = wf;
    }
}

// ---------------------------------------------------------------------------
// Up+gate: h[t,:] = silu(x@W1[e]) * (x@W3[e]) for this block's 64 tokens and
// 128 F-columns. Mtile=64 (4 A-frags per B-frag), double-buffered transposed
// weight tiles, h written to workspace as bf16.
// grid = (F/128, T/64, E), 256 threads (8 waves).
// All global accesses: uniform base + 32-bit offset (SADDR-form loads).
// ---------------------------------------------------------------------------
__global__ __launch_bounds__(256) void moe_up(
    const float* __restrict__ x, const float* __restrict__ W1,
    const float* __restrict__ W3,
    const int* __restrict__ counts, const int* __restrict__ list,
    unsigned short* __restrict__ h_ws) {

    extern __shared__ unsigned char smem[];
    unsigned short* a_lds = (unsigned short*)smem;          // 64 x 512 bf16 (64KB)
    unsigned short* wlds  = a_lds + 64 * Dd;                // [2 buf][2 w][128*WP]
    int* tok = (int*)(wlds + 2 * 2 * 128 * WP);             // 64

    const int jbase = blockIdx.x * 128;
    const int mt    = blockIdx.y;
    const int e     = blockIdx.z;
    const int cnt   = counts[e];
    const int m0    = mt * 64;
    if (m0 >= cnt) return;                                  // uniform exit
    const int mvalid = (cnt - m0 < 64) ? (cnt - m0) : 64;

    const int tid = threadIdx.x;
    if (tid < 64) tok[tid] = (tid < mvalid) ? list[e * Tt + m0 + tid] : 0;
    __syncthreads();

    // Stage A: 64 x 512 f32 -> bf16. Per iteration all 256 threads read one
    // token row (2KB contiguous); LDS store offsets are immediates.
    {
        unsigned* a32 = (unsigned*)a_lds;
#pragma unroll 4
        for (int r = 0; r < 64; ++r) {
            unsigned off = (unsigned)tok[r] * (unsigned)Dd + 2u * (unsigned)tid;
            const float2 v = *(const float2*)(x + off);
            a32[r * 256 + tid] = pkbf(v.x, v.y);
        }
    }

    const float* W1e = W1 + (size_t)e * Dd * Ff + jbase;    // uniform bases
    const float* W3e = W3 + (size_t)e * Dd * Ff + jbase;

    // Stage one 32(K) x 128(N) chunk of W1 and W3, transposed to [n][k] pairs.
    // n fixed per thread, kp affine -> global/LDS offsets fold to immediates.
    auto stage_w = [&](int kc, int buf) {
        const int n   = tid & 127;
        const int kp0 = tid >> 7;                      // 0..1
        unsigned  off = (unsigned)(kc + 2 * kp0) * (unsigned)Ff + (unsigned)n;
        unsigned* d1  = (unsigned*)(wlds + (size_t)(buf * 2 + 0) * 128 * WP)
                        + n * (WP / 2) + kp0;
        unsigned* d3  = (unsigned*)(wlds + (size_t)(buf * 2 + 1) * 128 * WP)
                        + n * (WP / 2) + kp0;
#pragma unroll
        for (int i = 0; i < 8; ++i) {
            float a0 = W1e[off], a1 = W1e[off + (unsigned)Ff];
            float c0 = W3e[off], c1 = W3e[off + (unsigned)Ff];
            d1[2 * i] = pkbf(a0, a1);
            d3[2 * i] = pkbf(c0, c1);
            off += 4u * (unsigned)Ff;
        }
    };

    stage_w(0, 0);
    __syncthreads();                           // covers A + first W tiles

    const int wave = tid >> 5, lane = tid & 31;
    const int nlo  = lane & 15, hi = lane >> 4;

    v8f acc1[4] = {vzero(), vzero(), vzero(), vzero()};
    v8f acc3[4] = {vzero(), vzero(), vzero(), vzero()};

#pragma unroll 1
    for (int kk = 0; kk < Dd / 32; ++kk) {
        if (kk + 1 < Dd / 32) {
            stage_w(32 * (kk + 1), (kk + 1) & 1);      // overlap with WMMA below
            __builtin_prefetch(W1e + (unsigned)(32 * (kk + 1)) * (unsigned)Ff + (tid & 127), 0, 1);
        }
        const unsigned short* wb =
            wlds + (size_t)((kk & 1) * 2) * 128 * WP + (wave * 16 + nlo) * WP + hi * 8;
        v16bf b1 = load_frag_b(wb);
        v16bf b3 = load_frag_b(wb + 128 * WP);
#pragma unroll
        for (int m = 0; m < 4; ++m) {
            const unsigned short* ap = a_lds + (m * 16 + nlo) * Dd + kk * 32 + hi * 8;
            v16bf a = load_frag_a(ap);
            acc1[m] = __builtin_amdgcn_wmma_f32_16x16x32_bf16(false, a, false, b1,
                                                              (short)0, acc1[m], false, false);
            acc3[m] = __builtin_amdgcn_wmma_f32_16x16x32_bf16(false, a, false, b3,
                                                              (short)0, acc3[m], false, false);
        }
        __syncthreads();
    }

    // SwiGLU, scatter h (bf16) to workspace by token id (32-bit offsets)
#pragma unroll
    for (int m = 0; m < 4; ++m) {
#pragma unroll
        for (int r = 0; r < 8; ++r) {
            int mrow = m * 16 + 8 * hi + r;
            if (mrow < mvalid) {
                float u = acc1[m][r];
                float g = acc3[m][r];
                float hv = (u * (1.0f / (1.0f + __expf(-u)))) * g;
                unsigned off = (unsigned)tok[mrow] * (unsigned)Ff
                             + (unsigned)(jbase + wave * 16 + nlo);
                h_ws[off] = f2bf(hv);
            }
        }
    }
}

// ---------------------------------------------------------------------------
// Down: out[t,:] = w_t * (h[t,:] @ W2[e]) for 64 tokens x 128 D-columns.
// A-tiles (already bf16 in ws) staged via global_load_async_to_lds_b128
// (ASYNCcnt), W2 tiles converted+transposed; both double-buffered.
// grid = (D/128, T/64, E), 256 threads.
// ---------------------------------------------------------------------------
__global__ __launch_bounds__(256) void moe_down(
    const unsigned short* __restrict__ h_ws, const float* __restrict__ W2,
    const int* __restrict__ counts, const int* __restrict__ list,
    const float* __restrict__ wlist, float* __restrict__ out) {

    __shared__ unsigned short a_sh[2 * 64 * WP];     // 2 x (64 rows x 32k +pad)
    __shared__ unsigned short w_sh[2 * 128 * WP];    // 2 x (128 n x 32k +pad)
    __shared__ int   tok[64];
    __shared__ float twt[64];

    const int jbase = blockIdx.x * 128;
    const int mt    = blockIdx.y;
    const int e     = blockIdx.z;
    const int cnt   = counts[e];
    const int m0    = mt * 64;
    if (m0 >= cnt) return;
    const int mvalid = (cnt - m0 < 64) ? (cnt - m0) : 64;

    const int tid = threadIdx.x;
    if (tid < 64) {
        tok[tid] = (tid < mvalid) ? list[e * Tt + m0 + tid] : 0;
        twt[tid] = (tid < mvalid) ? wlist[e * Tt + m0 + tid] : 0.0f;
    }
    __syncthreads();

    const unsigned a_base = (unsigned)(size_t)&a_sh[0];   // LDS byte offset (addr[31:0])

    // Async copy: one b128 per lane; 64 rows x 64B, bf16 already -> pure bytes.
    auto stage_a = [&](int kc, int buf) {
        int r = tid >> 2, q = tid & 3;
        const unsigned short* gp =
            h_ws + ((unsigned)tok[r] * (unsigned)Ff + (unsigned)(kc + q * 8));
        unsigned ldsoff = a_base + (unsigned)(buf * 64 * WP * 2 + r * WP * 2 + q * 16);
        unsigned long long ga = (unsigned long long)(size_t)gp;
        asm volatile("global_load_async_to_lds_b128 %0, %1, off"
                     :: "v"(ldsoff), "v"(ga) : "memory");
    };

    const float* W2e = W2 + (size_t)e * Ff * Dd + jbase;
    auto stage_w2 = [&](int kc, int buf) {
        const int n   = tid & 127;
        const int kp0 = tid >> 7;
        unsigned  off = (unsigned)(kc + 2 * kp0) * (unsigned)Dd + (unsigned)n;
        unsigned* wt  = (unsigned*)(w_sh + (size_t)buf * 128 * WP) + n * (WP / 2) + kp0;
#pragma unroll
        for (int i = 0; i < 8; ++i) {
            wt[2 * i] = pkbf(W2e[off], W2e[off + (unsigned)Dd]);
            off += 4u * (unsigned)Dd;
        }
    };

    stage_a(0, 0);
    stage_w2(0, 0);
    asm volatile("s_wait_asynccnt 0" ::: "memory");
    __syncthreads();

    const int wave = tid >> 5, lane = tid & 31;
    const int nlo  = lane & 15, hi = lane >> 4;

    v8f acc[4] = {vzero(), vzero(), vzero(), vzero()};

#pragma unroll 1
    for (int kk = 0; kk < Ff / 32; ++kk) {
        int cur = kk & 1;
        if (kk + 1 < Ff / 32) {
            stage_a(32 * (kk + 1), cur ^ 1);
            stage_w2(32 * (kk + 1), cur ^ 1);
            __builtin_prefetch(W2e + (unsigned)(32 * (kk + 2)) * (unsigned)Dd + (tid & 127), 0, 1);
        }
        const unsigned short* wb = w_sh + (size_t)cur * 128 * WP + (wave * 16 + nlo) * WP + hi * 8;
        v16bf b = load_frag_b(wb);
#pragma unroll
        for (int m = 0; m < 4; ++m) {
            const unsigned short* ap = a_sh + (size_t)cur * 64 * WP + (m * 16 + nlo) * WP + hi * 8;
            v16bf a = load_frag_a(ap);
            acc[m] = __builtin_amdgcn_wmma_f32_16x16x32_bf16(false, a, false, b,
                                                             (short)0, acc[m], false, false);
        }
        asm volatile("s_wait_asynccnt 0" ::: "memory");
        __syncthreads();
    }

    // out[t, jbase + wave*16 + nlo] = w_t * y  (32-bit offsets)
#pragma unroll
    for (int m = 0; m < 4; ++m) {
#pragma unroll
        for (int r = 0; r < 8; ++r) {
            int mrow = m * 16 + 8 * hi + r;
            if (mrow < mvalid) {
                unsigned off = (unsigned)tok[mrow] * (unsigned)Dd
                             + (unsigned)(jbase + wave * 16 + nlo);
                out[off] = twt[mrow] * acc[m][r];
            }
        }
    }
}

// ---------------------------------------------------------------------------
extern "C" void kernel_launch(void* const* d_in, const int* in_sizes, int n_in,
                              void* d_out, int out_size, void* d_ws, size_t ws_size,
                              hipStream_t stream) {
    (void)in_sizes; (void)n_in; (void)out_size; (void)ws_size;

    const float* x  = (const float*)d_in[0];
    const float* Wg = (const float*)d_in[1];
    const float* W1 = (const float*)d_in[2];
    const float* W2 = (const float*)d_in[3];
    const float* W3 = (const float*)d_in[4];
    float* out = (float*)d_out;

    // ws: [counts 256B] [list E*T int] [wlist E*T f32] [h T*F bf16 (8MB)]
    char* ws = (char*)d_ws;
    int*            counts = (int*)ws;
    int*            list   = (int*)(ws + 256);
    float*          wlist  = (float*)(ws + 256 + (size_t)Ee * Tt * sizeof(int));
    unsigned short* h_ws   = (unsigned short*)(ws + 256 + (size_t)Ee * Tt * 8);

    moe_zero<<<1, 32, 0, stream>>>(counts);
    moe_route<<<(Tt * 32) / 256, 256, 0, stream>>>(x, Wg, counts, list, wlist);

    const size_t smem_up = (size_t)(64 * Dd + 2 * 2 * 128 * WP) * sizeof(unsigned short)
                           + 64 * sizeof(int);   // ~104.3 KB < 320 KB/WGP (3 blocks/WGP)
    (void)hipFuncSetAttribute(reinterpret_cast<const void*>(moe_up),
                              hipFuncAttributeMaxDynamicSharedMemorySize, (int)smem_up);

    moe_up<<<dim3(Ff / 128, Tt / 64, Ee), 256, smem_up, stream>>>(
        x, W1, W3, counts, list, h_ws);

    moe_down<<<dim3(Dd / 128, Tt / 64, Ee), 256, 0, stream>>>(
        h_ws, W2, counts, list, wlist, out);
}
